// RNN_LSTM_attention_one_step_90907277787354
// MI455X (gfx1250) — compile-verified, matrix operand order checked by reading, not run on hardware
//
#include <hip/hip_runtime.h>
#include <math.h>

// Problem dims (from reference): B=32, T=256, I=128, H=128, C=100
#define Bq 32
#define Tq 256
#define Iq 128
#define Hq 128
#define G4q 512   // 4*H
#define Cq 100

typedef float  v2f  __attribute__((ext_vector_type(2)));
typedef float  v8f  __attribute__((ext_vector_type(8)));
typedef __bf16 v16bf __attribute__((ext_vector_type(16)));

__device__ __forceinline__ float sigf(float x) { return 1.0f / (1.0f + __expf(-x)); }

// CDNA5 async global->LDS copy, 16 bytes per lane (ASYNCcnt-tracked).
__device__ __forceinline__ void async_copy_b128(unsigned lds_off, const void* gaddr) {
  asm volatile("global_load_async_to_lds_b128 %0, %1, off"
               :: "v"(lds_off), "v"(gaddr) : "memory");
}
__device__ __forceinline__ void wait_asynccnt0() {
  asm volatile("s_wait_asynccnt 0x0" ::: "memory");
}

// ---------------------------------------------------------------------------
// Kernel A: pre-gates G[t][b][n] = x[b,t,:]@w_ih^T[.,n] + b_ih[n] + b_hh[n]
// fp32 WMMA 16x16x4. M = B*T = 8192 (row m = b*T+t), N = 512, K = 128.
// ---------------------------------------------------------------------------
__global__ __launch_bounds__(256)
void pregate_gemm(const float* __restrict__ X,
                  const float* __restrict__ w_ih,
                  const float* __restrict__ b_ih,
                  const float* __restrict__ b_hh,
                  float* __restrict__ Gt) {
  const int wave = threadIdx.x >> 5;
  const int lane = threadIdx.x & 31;
  const int tile = blockIdx.x * 8 + wave;   // 16384 tiles total
  const int mt = tile >> 5;                 // 0..511
  const int nt = tile & 31;                 // 0..31
  const int n  = nt * 16 + (lane & 15);
  const int hi = lane >> 4;
  const int mrow = mt * 16 + (lane & 15);   // A-frag row (lanes 16-31 reuse M, hold K=2,3)

  const float bias = b_ih[n] + b_hh[n];
  v8f acc;
#pragma unroll
  for (int v = 0; v < 8; ++v) acc[v] = bias;

  for (int k = 0; k < Iq; k += 4) {
    const int kk = k + hi * 2;
    v2f a, b;
    a.x = X[mrow * Iq + kk];     a.y = X[mrow * Iq + kk + 1];
    b.x = w_ih[n * Iq + kk];     b.y = w_ih[n * Iq + kk + 1];   // B[k][n] = w_ih[n][k]
    acc = __builtin_amdgcn_wmma_f32_16x16x4_f32(false, a, false, b, (short)0, acc, false, false);
  }
#pragma unroll
  for (int v = 0; v < 8; ++v) {
    const int m  = mt * 16 + v + hi * 8;    // global row in [B*T]
    const int bb = m >> 8;                  // T == 256
    const int tt = m & 255;
    Gt[(tt * Bq + bb) * G4q + n] = acc[v];  // store [t][b][512] for contiguous step reads
  }
}

// ---------------------------------------------------------------------------
// Kernel B: sequential LSTM, one workgroup (512 thr = 16 waves).
// w_hh resident in LDS as pre-swizzled bf16 WMMA B-fragments; h as A-fragments.
// Pre-gates for step t+1 are async-prefetched (GLOBAL_LOAD_ASYNC_TO_LDS_B128)
// into an LDS staging buffer while step t's activations and step t+1's WMMAs
// run; consumed after s_wait_asynccnt 0 + barrier.
// LDS: 128K (whhB) + 8K (hA) + 64K (gates) + 64K (Gbuf) = 264K <= 320K/WGP.
// ---------------------------------------------------------------------------
__global__ __launch_bounds__(512)
void lstm_seq(const float* __restrict__ Gt,
              const float* __restrict__ w_hh,  // [4H][H]
              float* __restrict__ Hmem) {      // [T][B][H], h^{(t+1)} at slot t
  extern __shared__ char smem[];
  __bf16* whhB = (__bf16*)smem;                     // [nt=32][kt=4][lane=32][e=16]  128 KB
  __bf16* hA   = (__bf16*)(smem + 131072);          // [mt=2][kt=4][lane=32][e=16]     8 KB
  float*  gsh  = (float*)(smem + 131072 + 8192);    // [32][512]                      64 KB
  float*  Gbuf = (float*)(smem + 131072 + 8192 + 65536);  // [32][512] prefetch       64 KB

  const int tid  = threadIdx.x;
  const int lane = tid & 31;
  const int wave = tid >> 5;

  // One-time: swizzle w_hh into bf16 B-fragments. B[k][n] = w_hh[n][k].
  // 16-bit fragment element e (VGPR v=e>>1, half sub=e&1) holds
  // k_local = (v&3)*2 + sub + (lane>>4)*8 + (v>>2)*16.
  for (int idx = tid; idx < 32 * 4 * 32 * 16; idx += 512) {
    const int e  = idx & 15;
    const int ln = (idx >> 4) & 31;
    const int kt = (idx >> 9) & 3;
    const int nt = idx >> 11;
    const int v = e >> 1, sub = e & 1;
    const int klocal = (v & 3) * 2 + sub + ((ln >> 4) & 1) * 8 + (v >> 2) * 16;
    const int k  = kt * 32 + klocal;
    const int nn = nt * 16 + (ln & 15);
    whhB[idx] = (__bf16)w_hh[nn * Hq + k];
  }
  for (int idx = tid; idx < 2 * 4 * 32 * 16; idx += 512) hA[idx] = (__bf16)0.0f;  // h0 = 0

  // LDS byte offset of Gbuf (low 32 bits of the flat shared pointer).
  const unsigned gbuf_lds = (unsigned)(uintptr_t)(void*)Gbuf;

  // Prologue: async-prefetch step 0's pre-gate slice (64 KB, 8 x b128/lane).
#pragma unroll
  for (int i = 0; i < 8; ++i) {
    const int c = tid + i * 512;                 // 16-byte chunk id, 4096 total
    async_copy_b128(gbuf_lds + c * 16, (const char*)Gt + (size_t)c * 16);
  }

  const int b     = tid >> 4;           // 0..31
  const int jbase = (tid & 15) * 8;     // 8 hidden elems per thread
  float creg[8];
#pragma unroll
  for (int u = 0; u < 8; ++u) creg[u] = 0.0f;

  __syncthreads();

  for (int t = 0; t < Tq; ++t) {
    // ---- Phase 1: all 16 WMMAs (h @ w_hh^T), accumulators start at zero ----
    v8f acc[4];
#pragma unroll
    for (int tt = 0; tt < 4; ++tt) {
      const int tile = wave * 4 + tt;   // 0..63
      const int mt = tile >> 5;         // 0..1
      const int nt = tile & 31;         // 0..31
      acc[tt] = (v8f){};
#pragma unroll
      for (int kt = 0; kt < 4; ++kt) {   // K = 128 in 4 chunks of 32
        v16bf a  = *(const v16bf*)&hA[((mt * 4 + kt) * 32 + lane) * 16];
        v16bf bm = *(const v16bf*)&whhB[((nt * 4 + kt) * 32 + lane) * 16];
        acc[tt] = __builtin_amdgcn_wmma_f32_16x16x32_bf16(false, a, false, bm,
                                                          (short)0, acc[tt], false, false);
      }
    }

    // ---- Phase 2: prefetched pre-gates are needed now ----
    wait_asynccnt0();      // own async loads landed in LDS
    __syncthreads();       // everyone's landed -> Gbuf fully valid
#pragma unroll
    for (int tt = 0; tt < 4; ++tt) {
      const int tile = wave * 4 + tt;
      const int mt = tile >> 5;
      const int nt = tile & 31;
      const int n  = nt * 16 + (lane & 15);
      const int hi = lane >> 4;
#pragma unroll
      for (int v = 0; v < 8; ++v) {
        const int bb = mt * 16 + v + hi * 8;
        gsh[bb * G4q + n] = acc[tt][v] + Gbuf[bb * G4q + n];
      }
    }
    __syncthreads();       // gsh ready; all Gbuf reads done -> safe to refill

    // ---- Phase 3: kick off async prefetch of step t+1 (uniform branch) ----
    if (t + 1 < Tq) {
      const char* src = (const char*)(Gt + (size_t)(t + 1) * Bq * G4q);
#pragma unroll
      for (int i = 0; i < 8; ++i) {
        const int c = tid + i * 512;
        async_copy_b128(gbuf_lds + c * 16, src + (size_t)c * 16);
      }
    }

    // ---- Phase 4: activations (gate order i,f,g,o) + repack h fragments ----
#pragma unroll
    for (int u = 0; u < 8; ++u) {
      const int j = jbase + u;
      const float ig = gsh[b * G4q + j];
      const float fg = gsh[b * G4q + Hq + j];
      const float gg = gsh[b * G4q + 2 * Hq + j];
      const float og = gsh[b * G4q + 3 * Hq + j];
      const float c = sigf(fg) * creg[u] + sigf(ig) * tanhf(gg);
      creg[u] = c;
      const float h = sigf(og) * tanhf(c);
      Hmem[(t * Bq + b) * Hq + j] = h;
      // inverse of the fragment mapping: k_local = j&31 within chunk kt = j>>5
      const int kt = j >> 5, klocal = j & 31;
      const int hi2 = (klocal >> 3) & 1;
      const int v   = ((klocal >> 1) & 3) + ((klocal >> 4) & 1) * 4;
      const int sub = klocal & 1;
      const int ln  = (b & 15) + hi2 * 16;
      const int mt  = b >> 4;
      hA[((mt * 4 + kt) * 32 + ln) * 16 + v * 2 + sub] = (__bf16)h;
    }
    __syncthreads();       // hA ready for next step's WMMAs
  }
}

// ---------------------------------------------------------------------------
// Kernel C1: V[j*32+b][k'] = h^{(j+1)}[b,:] @ w1_bot  (w1_bot[m][k']=w1[(H+m)][k'])
// fp32 WMMA. M=8192, N=128, K=128.
// ---------------------------------------------------------------------------
__global__ __launch_bounds__(256)
void attnv_gemm(const float* __restrict__ Hmem,
                const float* __restrict__ w1,   // [2H][H]
                float* __restrict__ V) {
  const int wave = threadIdx.x >> 5;
  const int lane = threadIdx.x & 31;
  const int tile = blockIdx.x * 8 + wave;  // 4096 tiles
  const int mt = tile >> 3;                // 0..511
  const int nt = tile & 7;                 // 0..7
  const int n  = nt * 16 + (lane & 15);
  const int hi = lane >> 4;
  const int mrow = mt * 16 + (lane & 15);

  v8f acc = {};
  for (int k = 0; k < Hq; k += 4) {
    const int kk = k + hi * 2;
    v2f a, b;
    a.x = Hmem[mrow * Hq + kk];      a.y = Hmem[mrow * Hq + kk + 1];
    b.x = w1[(Hq + kk) * Hq + n];    b.y = w1[(Hq + kk + 1) * Hq + n];
    acc = __builtin_amdgcn_wmma_f32_16x16x4_f32(false, a, false, b, (short)0, acc, false, false);
  }
#pragma unroll
  for (int v = 0; v < 8; ++v) {
    const int m = mt * 16 + v + hi * 8;
    V[m * Hq + n] = acc[v];
  }
}

// ---------------------------------------------------------------------------
// Kernel C2: per-batch final attention + FC.
// u = h_T @ w1_top; s_j = tanh(u + V_j)·w2; attn = sum_j s_j h_j (j=0..T-2);
// out = (h_T + attn) @ w_fc^T + b_fc.
// ---------------------------------------------------------------------------
__global__ __launch_bounds__(256)
void attn_fc(const float* __restrict__ Hmem,
             const float* __restrict__ V,
             const float* __restrict__ w1,
             const float* __restrict__ w2,
             const float* __restrict__ w_fc,
             const float* __restrict__ b_fc,
             float* __restrict__ out) {
  __shared__ float ht_s[Hq];
  __shared__ float u_s[Hq];
  __shared__ float s_s[Tq];
  __shared__ float attn_s[Hq];
  const int b = blockIdx.x;
  const int tid = threadIdx.x;

  if (tid < Hq) ht_s[tid] = Hmem[((Tq - 1) * Bq + b) * Hq + tid];
  __syncthreads();
  if (tid < Hq) {
    float u = 0.0f;
    for (int m = 0; m < Hq; ++m) u += ht_s[m] * w1[m * Hq + tid];
    u_s[tid] = u;
  }
  __syncthreads();
  {
    const int j = tid;           // 0..255; only j < T-1 contributes
    float s = 0.0f;
    if (j < Tq - 1) {
      const float* vr = &V[(j * Bq + b) * Hq];
      for (int k = 0; k < Hq; ++k) s += tanhf(u_s[k] + vr[k]) * w2[k];
    }
    s_s[j] = s;
  }
  __syncthreads();
  if (tid < Hq) {
    float a = 0.0f;
    for (int j = 0; j < Tq - 1; ++j) a += s_s[j] * Hmem[(j * Bq + b) * Hq + tid];
    attn_s[tid] = a;
  }
  __syncthreads();
  if (tid < Cq) {
    float o = b_fc[tid];
    for (int h = 0; h < Hq; ++h) o += (ht_s[h] + attn_s[h]) * w_fc[tid * Hq + h];
    out[b * Cq + tid] = o;
  }
}

// ---------------------------------------------------------------------------
extern "C" void kernel_launch(void* const* d_in, const int* in_sizes, int n_in,
                              void* d_out, int out_size, void* d_ws, size_t ws_size,
                              hipStream_t stream) {
  (void)in_sizes; (void)n_in; (void)out_size; (void)ws_size;
  const float* x    = (const float*)d_in[0];
  const float* w_ih = (const float*)d_in[1];
  const float* b_ih = (const float*)d_in[2];
  const float* w_hh = (const float*)d_in[3];
  const float* b_hh = (const float*)d_in[4];
  const float* w1   = (const float*)d_in[5];
  const float* w2   = (const float*)d_in[6];
  const float* w_fc = (const float*)d_in[7];
  const float* b_fc = (const float*)d_in[8];
  float* out = (float*)d_out;

  char* ws = (char*)d_ws;
  float* Gt   = (float*)ws;                                         // 16 MB: [T][B][512]
  float* Hmem = (float*)(ws + (size_t)Tq * Bq * G4q * sizeof(float));        // 4 MB
  float* V    = (float*)(ws + (size_t)Tq * Bq * G4q * sizeof(float)
                            + (size_t)Tq * Bq * Hq  * sizeof(float));        // 4 MB

  pregate_gemm<<<2048, 256, 0, stream>>>(x, w_ih, b_ih, b_hh, Gt);
  const size_t lds_bytes = 131072 + 8192 + 65536 + 65536;  // 264 KB of 320 KB WGP LDS
  lstm_seq<<<1, 512, lds_bytes, stream>>>(Gt, w_hh, Hmem);
  attnv_gemm<<<512, 256, 0, stream>>>(Hmem, w1, V);
  attn_fc<<<32, 256, 0, stream>>>(Hmem, V, w1, w2, w_fc, b_fc, out);
}
